// GPT2Attention_55387898249249
// MI455X (gfx1250) — matrix operand order repaired and checked by reading, not compiled, not run on hardware
//
#include <hip/hip_runtime.h>

#define SEQ     2048
#define DMODEL  1024
#define NHEAD   16
#define HDIM    64
#define N3      3072

typedef __attribute__((ext_vector_type(16))) _Float16 v16h;
typedef __attribute__((ext_vector_type(8)))  _Float16 v8h;
typedef __attribute__((ext_vector_type(4)))  _Float16 v4h;
typedef __attribute__((ext_vector_type(8)))  float    v8f;

// ---------------- fragment helpers (CDNA5 WMMA layouts, ISA 7.12.2) --------

static __device__ __forceinline__ v16h join16(v8h lo, v8h hi) {
  v16h r;
#pragma unroll
  for (int i = 0; i < 8; ++i) { r[i] = lo[i]; r[i + 8] = hi[i]; }
  return r;
}

// A fragment 16x32 f16 from row-major [row][k] memory (stride in halves):
//  lane<16 : row=lane,    halves K 0..7  and 16..23
//  lane>=16: row=lane-16, halves K 8..15 and 24..31
static __device__ __forceinline__ v16h load_a_frag(const _Float16* base, int stride, int lane) {
  const int row = lane & 15;
  const int ho  = (lane >> 4) << 3;                       // 0 or 8
  const _Float16* p = base + (size_t)row * stride + ho;
  return join16(*(const v8h*)p, *(const v8h*)(p + 16));
}

// B fragment 32x16 f16 from [n][k]-major memory (stride in halves):
//  lane<16 : col=lane,    K 0..7  and 16..23
//  lane>=16: col=lane-16, K 8..15 and 24..31
static __device__ __forceinline__ v16h load_b_frag(const _Float16* base, int stride, int lane) {
  const int row = lane & 15;
  const int ho  = (lane >> 4) << 3;                       // 0 or 8
  const _Float16* p = base + (size_t)row * stride + ho;
  return join16(*(const v8h*)p, *(const v8h*)(p + 16));
}

static __device__ __forceinline__ v8f wmma16(v16h a, v16h b, v8f c) {
  // D = A(16x32) * B(32x16) + C, f32 accumulate
  return __builtin_amdgcn_wmma_f32_16x16x32_f16(false, a, false, b, (short)0, c, false, false);
}

// XOR-shuffle reductions confined to each 16-lane half (ds_swizzle group-of-32,
// and_mask=0x1f, xor_mask=1/2/4/8)
static __device__ __forceinline__ float red16_max(float v) {
  v = fmaxf(v, __int_as_float(__builtin_amdgcn_ds_swizzle(__float_as_int(v), 0x041f)));
  v = fmaxf(v, __int_as_float(__builtin_amdgcn_ds_swizzle(__float_as_int(v), 0x081f)));
  v = fmaxf(v, __int_as_float(__builtin_amdgcn_ds_swizzle(__float_as_int(v), 0x101f)));
  v = fmaxf(v, __int_as_float(__builtin_amdgcn_ds_swizzle(__float_as_int(v), 0x201f)));
  return v;
}
static __device__ __forceinline__ float red16_sum(float v) {
  v += __int_as_float(__builtin_amdgcn_ds_swizzle(__float_as_int(v), 0x041f));
  v += __int_as_float(__builtin_amdgcn_ds_swizzle(__float_as_int(v), 0x081f));
  v += __int_as_float(__builtin_amdgcn_ds_swizzle(__float_as_int(v), 0x101f));
  v += __int_as_float(__builtin_amdgcn_ds_swizzle(__float_as_int(v), 0x201f));
  return v;
}

// ---------------- prep kernels ---------------------------------------------

__global__ __launch_bounds__(256) void cvt_f32_f16(const float* __restrict__ in,
                                                   _Float16* __restrict__ out) {
  const size_t i = ((size_t)blockIdx.x * blockDim.x + threadIdx.x) * 4;
  const float4 v = *(const float4*)(in + i);
  v4h o;
  o[0] = (_Float16)v.x; o[1] = (_Float16)v.y;
  o[2] = (_Float16)v.z; o[3] = (_Float16)v.w;
  *(v4h*)(out + i) = o;
}

// in: f32 [K][N]  ->  out: f16 [N][K]
__global__ __launch_bounds__(256) void transp_cvt(const float* __restrict__ in,
                                                  _Float16* __restrict__ out,
                                                  int K, int N) {
  __shared__ float tile[32][33];
  const int bx = blockIdx.x * 32;          // N
  const int by = blockIdx.y * 32;          // K
  const int tx = threadIdx.x & 31;
  const int ty = threadIdx.x >> 5;         // 0..7
#pragma unroll
  for (int i = 0; i < 32; i += 8)
    tile[ty + i][tx] = in[(size_t)(by + ty + i) * N + bx + tx];
  __syncthreads();
#pragma unroll
  for (int i = 0; i < 32; i += 8)
    out[(size_t)(bx + ty + i) * K + by + tx] = (_Float16)tile[tx][ty + i];
}

// ---------------- QKV GEMM: [2048,1024] x [1024,3072] + bias ---------------
// 8 waves/WG, wave = 32x64 tile; scatter q/k per-head row-major, v transposed.

__global__ __launch_bounds__(256) void qkv_gemm(const _Float16* __restrict__ Ah,
                                                const _Float16* __restrict__ Bt,
                                                const float* __restrict__ bias,
                                                _Float16* __restrict__ qh,
                                                _Float16* __restrict__ kh,
                                                _Float16* __restrict__ vt) {
  const int lane = threadIdx.x & 31;
  const int wave = threadIdx.x >> 5;
  const int mb = blockIdx.x * 128 + (wave & 3) * 32;
  const int nb = blockIdx.y * 128 + (wave >> 2) * 64;

  v8f acc[2][4];
#pragma unroll
  for (int i = 0; i < 2; ++i)
#pragma unroll
    for (int j = 0; j < 4; ++j)
#pragma unroll
      for (int r = 0; r < 8; ++r) acc[i][j][r] = 0.0f;

  for (int kt = 0; kt < DMODEL; kt += 32) {
    const v16h a0 = load_a_frag(Ah + (size_t)mb * DMODEL + kt, DMODEL, lane);
    const v16h a1 = load_a_frag(Ah + (size_t)(mb + 16) * DMODEL + kt, DMODEL, lane);
#pragma unroll
    for (int j = 0; j < 4; ++j) {
      const v16h b = load_b_frag(Bt + (size_t)(nb + j * 16) * DMODEL + kt, DMODEL, lane);
      acc[0][j] = wmma16(a0, b, acc[0][j]);
      acc[1][j] = wmma16(a1, b, acc[1][j]);
    }
  }

  const int nlo = lane & 15;
  const int hi8 = (lane >> 4) << 3;
#pragma unroll
  for (int j = 0; j < 4; ++j) {
    const int c  = nb + j * 16 + nlo;
    const float bv = bias[c];
    const int seg = c >> 10;            // 0=q 1=k 2=v
    const int hh  = (c & 1023) >> 6;
    const int d   = c & 63;
#pragma unroll
    for (int mi = 0; mi < 2; ++mi)
#pragma unroll
      for (int r = 0; r < 8; ++r) {
        const int s = mb + mi * 16 + r + hi8;
        const _Float16 hv = (_Float16)(acc[mi][j][r] + bv);
        if (seg == 0)      qh[((size_t)hh * SEQ + s) * HDIM + d] = hv;
        else if (seg == 1) kh[((size_t)hh * SEQ + s) * HDIM + d] = hv;
        else               vt[((size_t)hh * HDIM + d) * SEQ + s] = hv;
      }
  }
}

// ---------------- fused causal attention (flash-style, 1 wave / 16 queries) -

__global__ __launch_bounds__(32) void attn_kernel(const _Float16* __restrict__ qh,
                                                  const _Float16* __restrict__ kh,
                                                  const _Float16* __restrict__ vt,
                                                  _Float16* __restrict__ ah) {
  __shared__ __align__(16) _Float16 lds_p[2][16 * 32];
  const int lane = threadIdx.x & 31;
  const int h   = blockIdx.x >> 7;        // / (SEQ/16)
  const int qb  = blockIdx.x & 127;
  const int nlo = lane & 15;
  const int hi8 = (lane >> 4) << 3;

  const _Float16* qbase = qh + ((size_t)h * SEQ + (size_t)qb * 16) * HDIM;
  const _Float16* kbase = kh + (size_t)h * SEQ * HDIM;
  const _Float16* vbase = vt + (size_t)h * HDIM * SEQ;

  const v16h qa0 = load_a_frag(qbase + 0,  HDIM, lane);
  const v16h qa1 = load_a_frag(qbase + 32, HDIM, lane);

  v8f o[4];
  float m_r[8], l_r[8];
#pragma unroll
  for (int t = 0; t < 4; ++t)
#pragma unroll
    for (int r = 0; r < 8; ++r) o[t][r] = 0.0f;
#pragma unroll
  for (int r = 0; r < 8; ++r) { m_r[r] = -1e30f; l_r[r] = 0.0f; }

  const int kbmax = (qb * 16 + 15) >> 5;          // causal: keys <= last query
  for (int kb = 0; kb <= kbmax; ++kb) {
    const int k0 = kb * 32;
    v8f s0, s1;
#pragma unroll
    for (int r = 0; r < 8; ++r) { s0[r] = 0.0f; s1[r] = 0.0f; }

    // scores: Q(16x64) x K^T(64x32), no 1/sqrt(d) scaling (prun path)
    {
      const v16h b00 = load_b_frag(kbase + (size_t)(k0 +  0) * HDIM +  0, HDIM, lane);
      const v16h b01 = load_b_frag(kbase + (size_t)(k0 +  0) * HDIM + 32, HDIM, lane);
      s0 = wmma16(qa0, b00, s0);
      s0 = wmma16(qa1, b01, s0);
      const v16h b10 = load_b_frag(kbase + (size_t)(k0 + 16) * HDIM +  0, HDIM, lane);
      const v16h b11 = load_b_frag(kbase + (size_t)(k0 + 16) * HDIM + 32, HDIM, lane);
      s1 = wmma16(qa0, b10, s1);
      s1 = wmma16(qa1, b11, s1);
    }

    float p0[8], p1[8], scal[8];
#pragma unroll
    for (int r = 0; r < 8; ++r) {
      const int qi = qb * 16 + r + hi8;
      float w0 = (k0 + nlo      > qi) ? -10000.0f : s0[r];
      float w1 = (k0 + 16 + nlo > qi) ? -10000.0f : s1[r];
      // soft-threshold prune: c + (w-c)*sigmoid(s*w), c=-1e4, s=10, alpha=0
      w0 = -10000.0f + (w0 + 10000.0f) * (1.0f / (1.0f + __expf(-10.0f * w0)));
      w1 = -10000.0f + (w1 + 10000.0f) * (1.0f / (1.0f + __expf(-10.0f * w1)));
      const float rm = red16_max(fmaxf(w0, w1));
      const float mn = fmaxf(m_r[r], rm);
      const float sc = __expf(m_r[r] - mn);
      m_r[r]  = mn;
      scal[r] = sc;
      p0[r] = __expf(w0 - mn);
      p1[r] = __expf(w1 - mn);
      l_r[r] = l_r[r] * sc + red16_sum(p0[r] + p1[r]);
    }
#pragma unroll
    for (int t = 0; t < 4; ++t)
#pragma unroll
      for (int r = 0; r < 8; ++r) o[t][r] *= scal[r];

    // C-layout -> A-layout re-tile of P through LDS (double buffered)
    const int buf = kb & 1;
    __syncthreads();
#pragma unroll
    for (int r = 0; r < 8; ++r) {
      lds_p[buf][(r + hi8) * 32 + nlo]      = (_Float16)p0[r];
      lds_p[buf][(r + hi8) * 32 + 16 + nlo] = (_Float16)p1[r];
    }
    __syncthreads();
    const v16h pa = load_a_frag(&lds_p[buf][0], 32, lane);

    // O += P(16x32) x V(32x64); V stored [d][s] so B-frags are contiguous
#pragma unroll
    for (int t = 0; t < 4; ++t) {
      const v16h vb = load_b_frag(vbase + (size_t)(t * 16) * SEQ + k0, SEQ, lane);
      o[t] = wmma16(pa, vb, o[t]);
    }
  }

#pragma unroll
  for (int t = 0; t < 4; ++t)
#pragma unroll
    for (int r = 0; r < 8; ++r) {
      const int s = qb * 16 + r + hi8;
      ah[(size_t)s * DMODEL + h * HDIM + t * 16 + nlo] = (_Float16)(o[t][r] / l_r[r]);
    }
}

// ---------------- output projection: [2048,1024] x [1024,1024] + bias ------

__global__ __launch_bounds__(256) void proj_gemm(const _Float16* __restrict__ Ah,
                                                 const _Float16* __restrict__ Bt,
                                                 const float* __restrict__ bias,
                                                 float* __restrict__ out) {
  const int lane = threadIdx.x & 31;
  const int wave = threadIdx.x >> 5;
  const int mb = blockIdx.x * 128 + (wave & 3) * 32;
  const int nb = blockIdx.y * 128 + (wave >> 2) * 64;

  v8f acc[2][4];
#pragma unroll
  for (int i = 0; i < 2; ++i)
#pragma unroll
    for (int j = 0; j < 4; ++j)
#pragma unroll
      for (int r = 0; r < 8; ++r) acc[i][j][r] = 0.0f;

  for (int kt = 0; kt < DMODEL; kt += 32) {
    const v16h a0 = load_a_frag(Ah + (size_t)mb * DMODEL + kt, DMODEL, lane);
    const v16h a1 = load_a_frag(Ah + (size_t)(mb + 16) * DMODEL + kt, DMODEL, lane);
#pragma unroll
    for (int j = 0; j < 4; ++j) {
      const v16h b = load_b_frag(Bt + (size_t)(nb + j * 16) * DMODEL + kt, DMODEL, lane);
      acc[0][j] = wmma16(a0, b, acc[0][j]);
      acc[1][j] = wmma16(a1, b, acc[1][j]);
    }
  }

  const int nlo = lane & 15;
  const int hi8 = (lane >> 4) << 3;
#pragma unroll
  for (int j = 0; j < 4; ++j) {
    const int c = nb + j * 16 + nlo;
    const float bv = bias[c];
#pragma unroll
    for (int mi = 0; mi < 2; ++mi)
#pragma unroll
      for (int r = 0; r < 8; ++r) {
        const int s = mb + mi * 16 + r + hi8;
        out[(size_t)s * DMODEL + c] = acc[mi][j][r] + bv;
      }
  }
}

// ---------------- launch ----------------------------------------------------

extern "C" void kernel_launch(void* const* d_in, const int* in_sizes, int n_in,
                              void* d_out, int out_size, void* d_ws, size_t ws_size,
                              hipStream_t stream) {
  const float* hs    = (const float*)d_in[0];
  const float* wqkv  = (const float*)d_in[1];
  const float* bqkv  = (const float*)d_in[2];
  const float* wproj = (const float*)d_in[3];
  const float* bproj = (const float*)d_in[4];
  float* out = (float*)d_out;

  _Float16* p = (_Float16*)d_ws;
  _Float16* hs_h = p; p += (size_t)SEQ * DMODEL;        // f16 hidden states
  _Float16* wt   = p; p += (size_t)N3 * DMODEL;         // c_attn_w^T f16 [3072][1024]
  _Float16* pt   = p; p += (size_t)DMODEL * DMODEL;     // c_proj_w^T f16 [1024][1024]
  _Float16* qh   = p; p += (size_t)NHEAD * SEQ * HDIM;  // [h][s][64]
  _Float16* kh   = p; p += (size_t)NHEAD * SEQ * HDIM;  // [h][s][64]
  _Float16* vt   = p; p += (size_t)NHEAD * HDIM * SEQ;  // [h][64][s]
  _Float16* ah   = p; p += (size_t)SEQ * DMODEL;        // attention output f16

  cvt_f32_f16<<<(SEQ * DMODEL) / (256 * 4), 256, 0, stream>>>(hs, hs_h);
  transp_cvt<<<dim3(N3 / 32, DMODEL / 32), 256, 0, stream>>>(wqkv, wt, DMODEL, N3);
  transp_cvt<<<dim3(DMODEL / 32, DMODEL / 32), 256, 0, stream>>>(wproj, pt, DMODEL, DMODEL);
  qkv_gemm<<<dim3(SEQ / 128, N3 / 128), 256, 0, stream>>>(hs_h, wt, bqkv, qh, kh, vt);
  attn_kernel<<<NHEAD * (SEQ / 16), 32, 0, stream>>>(qh, kh, vt, ah);
  proj_gemm<<<dim3(SEQ / 128, DMODEL / 128), 256, 0, stream>>>(ah, pt, bproj, out);
}
